// MambaEncoder_83227876262103
// MI455X (gfx1250) — compile-verified, matrix operand order checked by reading, not run on hardware
//
#include <hip/hip_runtime.h>
#include <hip/hip_bf16.h>

// ---------------- problem constants ----------------
#define BATCH 4
#define LSEQ  4096
#define DMODEL 256          // DM
#define DINNER 512          // DI = 2*DM
#define E2    1024          // 2*DI (in_proj output)
#define DSTATE 16           // DS
#define DRANK  16           // DR
#define NDBC   48           // DR + 2*DS
#define DCONV  4
#define MROWS (BATCH*LSEQ)  // 16384
#define SCHUNK 16           // scan chunks over L
#define CLEN  (LSEQ/SCHUNK) // 256

typedef __bf16 v16bf __attribute__((ext_vector_type(16)));
typedef __bf16 v8bf  __attribute__((ext_vector_type(8)));
typedef float  v8f   __attribute__((ext_vector_type(8)));
typedef unsigned short v8u16 __attribute__((ext_vector_type(8)));
typedef unsigned int   v4u   __attribute__((ext_vector_type(4)));
typedef unsigned int   v8u   __attribute__((ext_vector_type(8)));

// ---------------- helpers ----------------
__device__ __forceinline__ unsigned short f2bf(float f) {
    unsigned int u = __float_as_uint(f);
    unsigned int r = u + 0x7FFFu + ((u >> 16) & 1u);   // round-to-nearest-even
    return (unsigned short)(r >> 16);
}
__device__ __forceinline__ float bf2f(unsigned short h) {
    return __uint_as_float(((unsigned int)h) << 16);
}
__device__ __forceinline__ float silu_f(float v) { return v / (1.0f + __expf(-v)); }
__device__ __forceinline__ float softplus_f(float v) { return (v > 20.0f) ? v : log1pf(__expf(v)); }

// 16x32 bf16 fragment loader (A-layout; also B for N-major weights).
// lane 0..15: row=lane, K=[0..7],[16..23]; lane 16..31: row=lane-16, K=[8..15],[24..31]
__device__ __forceinline__ v16bf load_frag(const unsigned short* base, int stride, int lane) {
    int r  = lane & 15;
    int kb = (lane & 16) ? 8 : 0;
    const unsigned short* p = base + r * stride + kb;
    v8u16 lo = *(const v8u16*)(p);
    v8u16 hi = *(const v8u16*)(p + 16);
    v8bf l2 = __builtin_bit_cast(v8bf, lo);
    v8bf h2 = __builtin_bit_cast(v8bf, hi);
    return __builtin_shufflevector(l2, h2, 0,1,2,3,4,5,6,7,8,9,10,11,12,13,14,15);
}

__device__ __forceinline__ v8f wmma_bf16(v16bf a, v16bf b, v8f c) {
    return __builtin_amdgcn_wmma_f32_16x16x32_bf16(false, a, false, b, (short)0, c, false, false);
}

// ---- Tensor Data Mover: 2-D bf16 tile (rows x width) -> LDS ----
// D# group0/group1 per CDNA5 ISA 8.3/8.4; groups 2/3 NULL (2-D tile).
// `lds_ptr` is passed into the asm so the LDS array ESCAPES: alias analysis
// must then treat the TDM as a writer of the tile (else the ds_loads fold to undef).
__device__ __forceinline__ void tdm_load_2d_bf16(const unsigned short* gptr,
                                                 void* lds_ptr, unsigned lds_byte_off,
                                                 unsigned width_elems, unsigned rows,
                                                 unsigned stride_elems) {
    unsigned long long ga = (unsigned long long)(__UINTPTR_TYPE__)gptr;
    v4u g0;
    g0[0] = 1u;                                        // count=1 (valid user D#)
    g0[1] = lds_byte_off;                              // lds_addr
    g0[2] = (unsigned)(ga & 0xffffffffu);              // global_addr[31:0]
    g0[3] = (unsigned)((ga >> 32) & 0x01ffffffu)       // global_addr[56:32]
          | 0x80000000u;                               // type=2 ("image")
    v8u g1;
    g1[0] = (1u << 16);                                // data_size=1 (2B), mask=0, no flags
    g1[1] = (width_elems & 0xffffu) << 16;             // tensor_dim0[15:0]
    g1[2] = ((width_elems >> 16) & 0xffffu)            // tensor_dim0[31:16]
          | ((rows & 0xffffu) << 16);                  // tensor_dim1[15:0]
    g1[3] = ((rows >> 16) & 0xffffu)                   // tensor_dim1[31:16]
          | ((width_elems & 0xffffu) << 16);           // tile_dim0
    g1[4] = (rows & 0xffffu);                          // tile_dim1 (tile_dim2=0)
    g1[5] = stride_elems;                              // tensor_dim0_stride[31:0]
    g1[6] = 0u;
    g1[7] = 0u;
    asm volatile("tensor_load_to_lds %0, %1"
                 :: "s"(g0), "s"(g1), "v"(lds_ptr)     // %2 unused: escapes LDS tile
                 : "memory");
}

// ---------------- K0: f32 -> bf16 weight conversion ----------------
__global__ void cvt_bf16_kernel(const float* __restrict__ in, unsigned short* __restrict__ out, int n) {
    int i = blockIdx.x * blockDim.x + threadIdx.x;
    if (i < n) out[i] = f2bf(in[i]);
}

// ---------------- K0b: wbar[d] = mean_e out_proj_w[e,d] ----------------
__global__ void wbar_kernel(const float* __restrict__ opw, float* __restrict__ wbar) {
    int d = blockIdx.x * blockDim.x + threadIdx.x;   // 0..511
    float s = 0.0f;
    for (int e = 0; e < DMODEL; ++e) s += opw[e * DINNER + d];
    wbar[d] = s * (1.0f / (float)DMODEL);
}

// ---------------- K1: conv3 + BN + SiLU -> h bf16 (B*L, DM) ----------------
__global__ void frontend_kernel(const float* __restrict__ x, const float* __restrict__ cw,
                                const float* __restrict__ gamma, const float* __restrict__ beta,
                                const float* __restrict__ mean,  const float* __restrict__ var,
                                unsigned short* __restrict__ hbf) {
    int bl = blockIdx.x;
    int d  = threadIdx.x;
    int l  = bl & (LSEQ - 1);
    int b  = bl >> 12;
    const float* xb = x + b * LSEQ;
    float xm = (l > 0)        ? xb[l - 1] : 0.0f;
    float x0 = xb[l];
    float xp = (l < LSEQ - 1) ? xb[l + 1] : 0.0f;
    float c  = xm * cw[d*3 + 0] + x0 * cw[d*3 + 1] + xp * cw[d*3 + 2];
    float inv = gamma[d] * rsqrtf(var[d] + 1e-5f);
    float v = (c - mean[d]) * inv + beta[d];
    hbf[bl * DMODEL + d] = f2bf(silu_f(v));
}

// ---------------- K2: in_proj GEMM (TDM B-tile -> LDS, WMMA 32x32/wave) ----------------
// xz[m, e] = sum_k h[m,k] * w[e,k];  M=16384, N=1024, K=256; bf16 out
__global__ void inproj_wmma_kernel(const unsigned short* __restrict__ hbf,
                                   const unsigned short* __restrict__ wbf,
                                   unsigned short* __restrict__ xzbf) {
    __shared__ __align__(16) unsigned short Btile[32 * DMODEL];   // 16 KB, LDS offset 0
    int tid  = threadIdx.x;
    int lane = tid & 31;
    int wv   = tid >> 5;
    int n32  = blockIdx.y;                 // 0..31
    int m32  = blockIdx.x * 8 + wv;        // 0..511

    if (tid == 0) {
        tdm_load_2d_bf16(wbf + n32 * 32 * DMODEL, Btile, /*lds_off=*/0u,
                         /*width=*/DMODEL, /*rows=*/32, /*stride=*/DMODEL);
        __builtin_amdgcn_s_wait_tensorcnt(0);
    }
    __syncthreads();

    const unsigned short* Abase = hbf + (m32 * 32) * DMODEL;
    v8f acc00 = {}, acc01 = {}, acc10 = {}, acc11 = {};
    for (int k = 0; k < DMODEL; k += 32) {
        if (k + 32 < DMODEL) __builtin_prefetch(Abase + k + 32, 0, 0);
        v16bf a0 = load_frag(Abase + k,             DMODEL, lane);
        v16bf a1 = load_frag(Abase + 16*DMODEL + k, DMODEL, lane);
        v16bf b0 = load_frag(Btile + k,             DMODEL, lane);     // ds_load_b128
        v16bf b1 = load_frag(Btile + 16*DMODEL + k, DMODEL, lane);
        acc00 = wmma_bf16(a0, b0, acc00);
        acc01 = wmma_bf16(a0, b1, acc01);
        acc10 = wmma_bf16(a1, b0, acc10);
        acc11 = wmma_bf16(a1, b1, acc11);
    }
    int nl = lane & 15;
    int mh = (lane >> 4) << 3;
    int m0 = m32 * 32, n0 = n32 * 32;
    #pragma unroll
    for (int r = 0; r < 8; ++r) {
        xzbf[(m0 + mh + r)      * E2 + n0 + nl]      = f2bf(acc00[r]);
        xzbf[(m0 + mh + r)      * E2 + n0 + 16 + nl] = f2bf(acc01[r]);
        xzbf[(m0 + 16 + mh + r) * E2 + n0 + nl]      = f2bf(acc10[r]);
        xzbf[(m0 + 16 + mh + r) * E2 + n0 + 16 + nl] = f2bf(acc11[r]);
    }
}

// ---------------- K3: depthwise causal conv4 + bias + SiLU ----------------
__global__ void dwconv_kernel(const unsigned short* __restrict__ xzbf, const float* __restrict__ c1w,
                              const float* __restrict__ c1b,
                              float* __restrict__ uc, unsigned short* __restrict__ ucbf) {
    int bl = blockIdx.x;
    int d  = threadIdx.x;              // 0..511
    int l  = bl & (LSEQ - 1);
    float acc = c1b[d];
    #pragma unroll
    for (int k = 0; k < DCONV; ++k) {
        int ll = l + k - (DCONV - 1);
        if (ll >= 0) acc += bf2f(xzbf[(bl + k - (DCONV - 1)) * E2 + d]) * c1w[d*DCONV + k];
    }
    float sv = silu_f(acc);
    uc[bl * DINNER + d]   = sv;
    ucbf[bl * DINNER + d] = f2bf(sv);
}

// ---------------- K4: x_proj GEMM (TDM whole weight -> LDS, WMMA 16x48/wave) ----------------
// dbc[m, j] = sum_k uc[m,k] * xw[j,k];  M=16384, N=48, K=512
__global__ void xproj_wmma_kernel(const unsigned short* __restrict__ ucbf,
                                  const unsigned short* __restrict__ xwbf,
                                  float* __restrict__ dbc) {
    __shared__ __align__(16) unsigned short Wt[NDBC * DINNER];   // 48 KB, LDS offset 0
    int tid  = threadIdx.x;
    int lane = tid & 31;
    int m16  = blockIdx.x * 8 + (tid >> 5);    // 0..1023

    if (tid == 0) {
        tdm_load_2d_bf16(xwbf, Wt, 0u, /*width=*/DINNER, /*rows=*/NDBC, /*stride=*/DINNER);
        __builtin_amdgcn_s_wait_tensorcnt(0);
    }
    __syncthreads();

    const unsigned short* Abase = ucbf + (m16 * 16) * DINNER;
    v8f acc0 = {}, acc1 = {}, acc2 = {};
    for (int k = 0; k < DINNER; k += 32) {
        if (k + 32 < DINNER) __builtin_prefetch(Abase + k + 32, 0, 0);
        v16bf a  = load_frag(Abase + k,              DINNER, lane);
        v16bf b0 = load_frag(Wt +  0 * DINNER + k,   DINNER, lane);
        v16bf b1 = load_frag(Wt + 16 * DINNER + k,   DINNER, lane);
        v16bf b2 = load_frag(Wt + 32 * DINNER + k,   DINNER, lane);
        acc0 = wmma_bf16(a, b0, acc0);
        acc1 = wmma_bf16(a, b1, acc1);
        acc2 = wmma_bf16(a, b2, acc2);
    }
    int nl = lane & 15;
    int mh = (lane >> 4) << 3;
    int m0 = m16 * 16;
    #pragma unroll
    for (int r = 0; r < 8; ++r) {
        dbc[(m0 + mh + r) * NDBC +  0 + nl] = acc0[r];
        dbc[(m0 + mh + r) * NDBC + 16 + nl] = acc1[r];
        dbc[(m0 + mh + r) * NDBC + 32 + nl] = acc2[r];
    }
}

// ---------------- K4b: dt = softplus(dbc[:, :16] @ dtw^T + b) ----------------
__global__ void dtproj_kernel(const float* __restrict__ dbc, const float* __restrict__ dtw,
                              const float* __restrict__ dtb, float* __restrict__ dt) {
    int idx = blockIdx.x * blockDim.x + threadIdx.x;   // m*512 + d
    int m = idx >> 9;
    int d = idx & (DINNER - 1);
    const float* row = dbc + m * NDBC;
    const float* w   = dtw + d * DRANK;
    float s = dtb[d];
    #pragma unroll
    for (int r = 0; r < DRANK; ++r) s += row[r] * w[r];
    dt[idx] = softplus_f(s);
}

// ---------------- K5: chunked selective scan ----------------
// Per-state recurrence h' = h*exp(A_s*dt) + dt*u*B is diagonal-linear:
// chunk contributes h_out = h_in * exp(A_s * sum(dt)) + local(h_in=0).
// S1: per-chunk carries (32768-way); S2: compose 16 carries/channel; S3: replay+emit.
__global__ void scan_part1(const float* __restrict__ dt, const float* __restrict__ uc,
                           const float* __restrict__ dbc, const float* __restrict__ A_log,
                           float* __restrict__ carry) {
    int idx = blockIdx.x * blockDim.x + threadIdx.x;   // b*8192 + c*512 + d
    int d = idx & (DINNER - 1);
    int c = (idx >> 9) & (SCHUNK - 1);
    int b = idx >> 13;
    float Arow[DSTATE];
    #pragma unroll
    for (int s = 0; s < DSTATE; ++s) Arow[s] = -__expf(A_log[d * DSTATE + s]);
    float h[DSTATE];
    #pragma unroll
    for (int s = 0; s < DSTATE; ++s) h[s] = 0.0f;
    float sumdt = 0.0f;
    int l0 = c * CLEN;
    for (int l = l0; l < l0 + CLEN; ++l) {
        int m = b * LSEQ + l;
        float dtv = dt[m * DINNER + d];
        float uv  = uc[m * DINNER + d];
        const float* row = dbc + m * NDBC;
        float du = dtv * uv;
        sumdt += dtv;
        #pragma unroll
        for (int s = 0; s < DSTATE; ++s) {
            float dA = __expf(dtv * Arow[s]);
            h[s] = h[s] * dA + du * row[DRANK + s];
        }
    }
    float* out = carry + (size_t)idx * 32;
    #pragma unroll
    for (int s = 0; s < DSTATE; ++s) {
        out[s]          = __expf(Arow[s] * sumdt);   // prod of dA over chunk
        out[DSTATE + s] = h[s];                      // chunk-local end state
    }
}

__global__ void scan_part2(const float* __restrict__ carry, float* __restrict__ h0buf) {
    int t = blockIdx.x * blockDim.x + threadIdx.x;   // b*DI + d (0..2047)
    int d = t & (DINNER - 1);
    int b = t >> 9;
    float h[DSTATE];
    #pragma unroll
    for (int s = 0; s < DSTATE; ++s) h[s] = 0.0f;
    for (int c = 0; c < SCHUNK; ++c) {
        size_t base = ((size_t)(b * SCHUNK + c) * DINNER + d);
        const float* cp = carry + base * 32;
        float* hp = h0buf + base * 16;
        #pragma unroll
        for (int s = 0; s < DSTATE; ++s) hp[s] = h[s];
        #pragma unroll
        for (int s = 0; s < DSTATE; ++s) h[s] = h[s] * cp[s] + cp[DSTATE + s];
    }
}

__global__ void scan_part3(const float* __restrict__ dt, const float* __restrict__ uc,
                           const float* __restrict__ dbc, const float* __restrict__ A_log,
                           const float* __restrict__ Dsk, const float* __restrict__ h0buf,
                           float* y_out) {
    int idx = blockIdx.x * blockDim.x + threadIdx.x;
    int d = idx & (DINNER - 1);
    int c = (idx >> 9) & (SCHUNK - 1);
    int b = idx >> 13;
    float Arow[DSTATE];
    #pragma unroll
    for (int s = 0; s < DSTATE; ++s) Arow[s] = -__expf(A_log[d * DSTATE + s]);
    float dskip = Dsk[d];
    const float* hp = h0buf + (size_t)idx * 16;
    float h[DSTATE];
    #pragma unroll
    for (int s = 0; s < DSTATE; ++s) h[s] = hp[s];
    int l0 = c * CLEN;
    for (int l = l0; l < l0 + CLEN; ++l) {
        int m = b * LSEQ + l;
        float dtv = dt[m * DINNER + d];
        float uv  = uc[m * DINNER + d];
        const float* row = dbc + m * NDBC;
        float du = dtv * uv;
        float y = 0.0f;
        #pragma unroll
        for (int s = 0; s < DSTATE; ++s) {
            float dA = __expf(dtv * Arow[s]);
            h[s] = h[s] * dA + du * row[DRANK + s];
            y += h[s] * row[DRANK + DSTATE + s];
        }
        y += uv * dskip;
        y_out[m * DINNER + d] = y;       // aliases dt buffer; same-element, read-before-write
    }
}

// ---------------- K6: out[m] = sum_d y*silu(z)*wbar ----------------
__global__ void reduce_out_kernel(const float* __restrict__ y, const unsigned short* __restrict__ xzbf,
                                  const float* __restrict__ wbar, float* __restrict__ out) {
    int gid  = blockIdx.x * blockDim.x + threadIdx.x;
    int m    = gid >> 5;
    int lane = threadIdx.x & 31;
    float s = 0.0f;
    for (int d = lane; d < DINNER; d += 32) {
        float z = bf2f(xzbf[m * E2 + DINNER + d]);
        s += y[m * DINNER + d] * silu_f(z) * wbar[d];
    }
    #pragma unroll
    for (int off = 16; off; off >>= 1) s += __shfl_xor(s, off, 32);
    if (lane == 0) out[m] = s;
}

// ---------------- host launcher ----------------
extern "C" void kernel_launch(void* const* d_in, const int* in_sizes, int n_in,
                              void* d_out, int out_size, void* d_ws, size_t ws_size,
                              hipStream_t stream) {
    const float* x        = (const float*)d_in[0];
    const float* conv_w   = (const float*)d_in[1];
    const float* bn_gamma = (const float*)d_in[2];
    const float* bn_beta  = (const float*)d_in[3];
    const float* bn_mean  = (const float*)d_in[4];
    const float* bn_var   = (const float*)d_in[5];
    const float* in_proj  = (const float*)d_in[6];
    const float* conv1d_w = (const float*)d_in[7];
    const float* conv1d_b = (const float*)d_in[8];
    const float* x_proj   = (const float*)d_in[9];
    const float* dt_projw = (const float*)d_in[10];
    const float* dt_projb = (const float*)d_in[11];
    const float* A_log    = (const float*)d_in[12];
    const float* D_skip   = (const float*)d_in[13];
    const float* out_proj = (const float*)d_in[14];
    float* out = (float*)d_out;

    char* ws = (char*)d_ws;
    size_t off = 0;
    auto take = [&](size_t bytes) { char* p = ws + off; off += (bytes + 255) & ~(size_t)255; return p; };
    unsigned short* hbf   = (unsigned short*)take((size_t)MROWS * DMODEL * 2);     //  8 MB
    unsigned short* winbf = (unsigned short*)take((size_t)E2 * DMODEL * 2);        //  0.5 MB
    unsigned short* xwbf  = (unsigned short*)take((size_t)NDBC * DINNER * 2);      //  48 KB
    float*          wbar  = (float*)take((size_t)DINNER * 4);                      //  2 KB
    unsigned short* xzbf  = (unsigned short*)take((size_t)MROWS * E2 * 2);         //  32 MB
    float*          uc    = (float*)take((size_t)MROWS * DINNER * 4);              //  32 MB
    unsigned short* ucbf  = (unsigned short*)take((size_t)MROWS * DINNER * 2);     //  16 MB
    float*          dbc   = (float*)take((size_t)MROWS * NDBC * 4);                //   3 MB
    float*          dtb   = (float*)take((size_t)MROWS * DINNER * 4);              //  32 MB
    float*          carry = (float*)take((size_t)BATCH * SCHUNK * DINNER * 32 * 4);//   4 MB
    float*          h0buf = (float*)take((size_t)BATCH * SCHUNK * DINNER * 16 * 4);//   2 MB
    float*          ybuf  = dtb;   // element-wise safe alias (read dt then write y)

    cvt_bf16_kernel<<<dim3((E2*DMODEL + 255)/256), dim3(256), 0, stream>>>(in_proj, winbf, E2*DMODEL);
    cvt_bf16_kernel<<<dim3((NDBC*DINNER + 255)/256), dim3(256), 0, stream>>>(x_proj, xwbf, NDBC*DINNER);
    wbar_kernel<<<dim3(2), dim3(256), 0, stream>>>(out_proj, wbar);

    frontend_kernel<<<dim3(MROWS), dim3(DMODEL), 0, stream>>>(x, conv_w, bn_gamma, bn_beta,
                                                              bn_mean, bn_var, hbf);

    // in_proj GEMM: grid (m-chunks, n-tiles); each block TDM-stages its 32x256 B tile
    inproj_wmma_kernel<<<dim3(64, 32), dim3(256), 0, stream>>>(hbf, winbf, xzbf);

    dwconv_kernel<<<dim3(MROWS), dim3(DINNER), 0, stream>>>(xzbf, conv1d_w, conv1d_b, uc, ucbf);

    // x_proj GEMM: each block TDM-stages the full 48x512 weight
    xproj_wmma_kernel<<<dim3(128), dim3(256), 0, stream>>>(ucbf, xwbf, dbc);

    dtproj_kernel<<<dim3((MROWS*DINNER)/256), dim3(256), 0, stream>>>(dbc, dt_projw, dt_projb, dtb);

    // chunked scan: S1 carries -> S2 compose -> S3 replay+emit
    scan_part1<<<dim3((BATCH*SCHUNK*DINNER)/256), dim3(256), 0, stream>>>(dtb, uc, dbc, A_log, carry);
    scan_part2<<<dim3((BATCH*DINNER)/256), dim3(256), 0, stream>>>(carry, h0buf);
    scan_part3<<<dim3((BATCH*SCHUNK*DINNER)/256), dim3(256), 0, stream>>>(dtb, uc, dbc, A_log, D_skip,
                                                                          h0buf, ybuf);

    reduce_out_kernel<<<dim3(MROWS/8), dim3(256), 0, stream>>>(ybuf, xzbf, wbar, out);

    (void)in_sizes; (void)n_in; (void)out_size; (void)ws_size;
}
